// TDRumorGCN_9182640079570
// MI455X (gfx1250) — matrix operand order repaired and checked by reading, not compiled, not run on hardware
//
#include <hip/hip_runtime.h>

#define N_NODES   200000
#define N_EDGES   200000
#define N_GRAPHS  128
#define IN_F      256
#define HID_F     128
#define OUT_F     128
#define OUTCAT_F  256   // OUT_F + HID_F (final concat width)

typedef __attribute__((ext_vector_type(16))) __bf16 v16bf;
typedef __attribute__((ext_vector_type(8)))  float  v8f;

// round-to-nearest-even f32 -> bf16
__device__ __forceinline__ __bf16 f2bf(float f) {
    unsigned u = __builtin_bit_cast(unsigned, f);
    unsigned r = (u + 0x7FFFu + ((u >> 16) & 1u)) >> 16;
    unsigned short s = (unsigned short)r;
    return __builtin_bit_cast(__bf16, s);
}

// ---------------------------------------------------------------- utilities
__global__ void fill_kernel(float* __restrict__ p, float v, int n) {
    int i = blockIdx.x * blockDim.x + threadIdx.x;
    if (i < n) p[i] = v;
}
__global__ void fill_int_kernel(int* __restrict__ p, int v, int n) {
    int i = blockIdx.x * blockDim.x + threadIdx.x;
    if (i < n) p[i] = v;
}

__global__ void indeg_kernel(const int* __restrict__ dst, int* __restrict__ cnt, int n_edges) {
    int e = blockIdx.x * blockDim.x + threadIdx.x;
    if (e < n_edges) atomicAdd(&cnt[dst[e]], 1);
}

__global__ void dinv_kernel(const int* __restrict__ cnt, float* __restrict__ dinv,
                            float* __restrict__ invdeg, int n) {
    int i = blockIdx.x * blockDim.x + threadIdx.x;
    if (i < n) {
        float d = 1.0f + (float)cnt[i];     // self-loop + in-degree
        dinv[i]   = rsqrtf(d);
        invdeg[i] = 1.0f / d;
    }
}

// ---------------------------------------------------------------- exclusive scan (3 kernels) for CSR row_ptr
__global__ __launch_bounds__(256) void scan_block_kernel(const int* __restrict__ cnt,
                                                         int* __restrict__ row_ptr,
                                                         int* __restrict__ blockSums, int n) {
    __shared__ int sh[256];
    int i = blockIdx.x * 256 + threadIdx.x;
    int v = (i < n) ? cnt[i] : 0;
    sh[threadIdx.x] = v;
    __syncthreads();
    for (int off = 1; off < 256; off <<= 1) {           // Hillis-Steele inclusive scan
        int t = (threadIdx.x >= off) ? sh[threadIdx.x - off] : 0;
        __syncthreads();
        sh[threadIdx.x] += t;
        __syncthreads();
    }
    if (i < n) row_ptr[i] = sh[threadIdx.x] - v;        // exclusive
    if (threadIdx.x == 255) blockSums[blockIdx.x] = sh[255];
}

__global__ __launch_bounds__(1024) void scan_sums_kernel(int* __restrict__ blockSums, int nb) {
    __shared__ int sh[1024];
    int t = threadIdx.x;
    int v = (t < nb) ? blockSums[t] : 0;
    sh[t] = v;
    __syncthreads();
    for (int off = 1; off < 1024; off <<= 1) {
        int x = (t >= off) ? sh[t - off] : 0;
        __syncthreads();
        sh[t] += x;
        __syncthreads();
    }
    if (t < nb) blockSums[t] = sh[t] - v;               // exclusive
}

__global__ void scan_add_kernel(int* __restrict__ row_ptr, const int* __restrict__ blockSums, int n) {
    int i = blockIdx.x * 256 + threadIdx.x;
    if (i < n) row_ptr[i] += blockSums[blockIdx.x];
}

// ---------------------------------------------------------------- bucket edges by destination (CSR column list)
__global__ void bucket_kernel(const int* __restrict__ src, const int* __restrict__ dst,
                              const int* __restrict__ row_ptr, int* __restrict__ cursor,
                              int* __restrict__ colsrc, int n_edges) {
    int e = blockIdx.x * blockDim.x + threadIdx.x;
    if (e >= n_edges) return;
    int d = dst[e];
    int pos = row_ptr[d] + atomicAdd(&cursor[d], 1);
    colsrc[pos] = src[e];
}

// Pack row-major W[K,128] (f32) into WMMA-B fragment order (bf16):
// t = ((kt*8 + nt)*32 + lane)*16 + j ; K = kt*32 + (lane>=16)*8 + (j&7) + (j>=8)*16 , N = nt*16 + (lane&15)
__global__ void pack_w_kernel(const float* __restrict__ W, __bf16* __restrict__ Wp, int ktiles) {
    int t = blockIdx.x * blockDim.x + threadIdx.x;
    if (t >= ktiles * 4096) return;
    int j    = t & 15;
    int lane = (t >> 4) & 31;
    int nt   = (t >> 9) & 7;
    int kt   = t >> 12;
    int K    = kt * 32 + (lane >> 4) * 8 + (j & 7) + (j >> 3) * 16;
    int Ncol = nt * 16 + (lane & 15);
    Wp[t] = f2bf(W[K * 128 + Ncol]);
}

// ---------------------------------------------------------------- GEMM1: hA = x @ W1  (bf16 WMMA, f32 acc)
__global__ __launch_bounds__(256) void gemm1_kernel(const float* __restrict__ x,
                                                    const __bf16* __restrict__ Wp,
                                                    float* __restrict__ hA) {
    __shared__ alignas(32) __bf16 Bs[8 * 32 * 16];       // one k-tile of 8 N-frags (8 KB)
    const int t    = threadIdx.x;
    const int lane = t & 31;
    const int wave = t >> 5;                             // 8 waves = 128 rows/block
    const int half = lane >> 4;
    const int l16  = lane & 15;
    const int rowBase = blockIdx.x * 128 + wave * 16;
    const int row  = rowBase + l16;
    const int arow = (row < N_NODES) ? row : 0;          // clamp (EXEC all-ones for WMMA)

    v8f acc[8];
#pragma unroll
    for (int i = 0; i < 8; ++i)
#pragma unroll
        for (int j = 0; j < 8; ++j) acc[i][j] = 0.0f;

    const float* xrow = x + (size_t)arow * IN_F;

    for (int kt = 0; kt < 8; ++kt) {
        __syncthreads();
        *(v16bf*)&Bs[t * 16] = *(const v16bf*)(Wp + (size_t)kt * 4096 + t * 16);
        if (kt < 7) __builtin_prefetch(Wp + (size_t)(kt + 1) * 4096 + t * 16, 0, 3);
        __syncthreads();

        // A-frag: 16-bit A 16x32 layout — lane<16: K 0..7 then 16..23; lane>=16: K 8..15 then 24..31
        v16bf a;
        const float* xp = xrow + kt * 32 + half * 8;
#pragma unroll
        for (int j = 0; j < 8; ++j) a[j]     = f2bf(xp[j]);
#pragma unroll
        for (int j = 0; j < 8; ++j) a[8 + j] = f2bf(xp[16 + j]);

#pragma unroll
        for (int nt = 0; nt < 8; ++nt) {
            v16bf b = *(const v16bf*)&Bs[(nt * 32 + lane) * 16];
            acc[nt] = __builtin_amdgcn_wmma_f32_16x16x32_bf16(
                false, a, false, b, (short)0, acc[nt], false, false);
        }
    }

    // C/D layout: VGPR j -> M = j + 8*half ; N = nt*16 + (lane&15)
#pragma unroll
    for (int nt = 0; nt < 8; ++nt)
#pragma unroll
        for (int j = 0; j < 8; ++j) {
            int r = rowBase + half * 8 + j;
            if (r < N_NODES) hA[(size_t)r * HID_F + nt * 16 + l16] = acc[nt][j];
        }
}

// ---------------------------------------------------------------- conv1 gather + self-loop + bias (no atomics)
// h1[i,k] = dinv[i] * sum_{e in CSR(i)} hA[src_e,k]*dinv[src_e] + hA[i,k]/deg[i] + b1[k]
__global__ __launch_bounds__(256) void gather_combine1_kernel(const float* __restrict__ hA,
                                                              const int* __restrict__ row_ptr,
                                                              const int* __restrict__ cnt,
                                                              const int* __restrict__ colsrc,
                                                              const float* __restrict__ dinv,
                                                              const float* __restrict__ invdeg,
                                                              const float* __restrict__ b1,
                                                              float* __restrict__ h1) {
    int gid = blockIdx.x * blockDim.x + threadIdx.x;
    if (gid >= N_NODES * HID_F) return;
    int i = gid >> 7;            // wave-uniform (32 lanes share one node's feature slice)
    int k = gid & 127;
    int s0 = row_ptr[i];
    int e0 = s0 + cnt[i];
    float acc = 0.0f;
    for (int e = s0; e < e0; ++e) {
        int s = colsrc[e];
        acc += hA[(size_t)s * HID_F + k] * dinv[s];
    }
    h1[gid] = acc * dinv[i] + hA[gid] * invdeg[i] + b1[k];
}

// ---------------------------------------------------------------- GEMM2: hB = relu(concat(h1, x[root])) @ W2
__global__ __launch_bounds__(256) void gemm2_kernel(const float* __restrict__ h1,
                                                    const float* __restrict__ x,
                                                    const int* __restrict__ batch,
                                                    const int* __restrict__ root_index,
                                                    const __bf16* __restrict__ Wp,
                                                    float* __restrict__ hB) {
    __shared__ alignas(32) __bf16 Bs[8 * 32 * 16];
    const int t    = threadIdx.x;
    const int lane = t & 31;
    const int wave = t >> 5;
    const int half = lane >> 4;
    const int l16  = lane & 15;
    const int rowBase = blockIdx.x * 128 + wave * 16;
    const int row  = rowBase + l16;
    const int arow = (row < N_NODES) ? row : 0;

    const int root = root_index[batch[arow]];
    const float* xr    = x  + (size_t)root * IN_F;    // 128 distinct root rows: L2-resident
    const float* h1row = h1 + (size_t)arow * HID_F;

    v8f acc[8];
#pragma unroll
    for (int i = 0; i < 8; ++i)
#pragma unroll
        for (int j = 0; j < 8; ++j) acc[i][j] = 0.0f;

    for (int kt = 0; kt < 12; ++kt) {                 // K = 384
        __syncthreads();
        *(v16bf*)&Bs[t * 16] = *(const v16bf*)(Wp + (size_t)kt * 4096 + t * 16);
        if (kt < 11) __builtin_prefetch(Wp + (size_t)(kt + 1) * 4096 + t * 16, 0, 3);
        __syncthreads();

        const float* base; int koff;
        if (kt < 4) { base = h1row; koff = kt * 32; }           // K 0..127   -> relu(h1)
        else        { base = xr;    koff = kt * 32 - HID_F; }   // K 128..383 -> relu(x[root])

        v16bf a;
        const float* ap = base + koff + half * 8;
#pragma unroll
        for (int j = 0; j < 8; ++j) a[j]     = f2bf(fmaxf(ap[j], 0.0f));
#pragma unroll
        for (int j = 0; j < 8; ++j) a[8 + j] = f2bf(fmaxf(ap[16 + j], 0.0f));

#pragma unroll
        for (int nt = 0; nt < 8; ++nt) {
            v16bf b = *(const v16bf*)&Bs[(nt * 32 + lane) * 16];
            acc[nt] = __builtin_amdgcn_wmma_f32_16x16x32_bf16(
                false, a, false, b, (short)0, acc[nt], false, false);
        }
    }

#pragma unroll
    for (int nt = 0; nt < 8; ++nt)
#pragma unroll
        for (int j = 0; j < 8; ++j) {
            int r = rowBase + half * 8 + j;
            if (r < N_NODES) hB[(size_t)r * OUT_F + nt * 16 + l16] = acc[nt][j];
        }
}

// ---------------------------------------------------------------- conv2 gather + relu + root-concat + segment sum
// Block = 256 threads (one per output feature) over 128 consecutive nodes.
// batch is sorted -> run-length accumulate, flush with atomics on graph change.
__global__ __launch_bounds__(256) void final_kernel(const float* __restrict__ hB,
                                                    const int* __restrict__ row_ptr,
                                                    const int* __restrict__ cnt,
                                                    const int* __restrict__ colsrc,
                                                    const float* __restrict__ dinv,
                                                    const float* __restrict__ invdeg,
                                                    const float* __restrict__ b2,
                                                    const float* __restrict__ h1,
                                                    const int* __restrict__ batch,
                                                    const int* __restrict__ root_index,
                                                    float* __restrict__ sums,
                                                    float* __restrict__ counts) {
    const int k  = threadIdx.x;                  // 0..255
    const int n0 = blockIdx.x * 128;
    const int nEnd = (n0 + 128 < N_NODES) ? n0 + 128 : N_NODES;

    float run = 0.0f, cnt_run = 0.0f;
    int gcur = batch[n0];
    for (int n = n0; n < nEnd; ++n) {
        int g = batch[n];
        if (g != gcur) {
            atomicAdd(&sums[(size_t)gcur * OUTCAT_F + k], run);
            if (k == 0) atomicAdd(&counts[gcur], cnt_run);
            run = 0.0f; cnt_run = 0.0f; gcur = g;
        }
        float v;
        if (k < OUT_F) {                          // relu(conv2): CSR gather, no atomics
            int s0 = row_ptr[n];
            int e0 = s0 + cnt[n];
            float acc = 0.0f;
            for (int e = s0; e < e0; ++e) {
                int s = colsrc[e];
                acc += hB[(size_t)s * OUT_F + k] * dinv[s];
            }
            v = acc * dinv[n] + hB[(size_t)n * OUT_F + k] * invdeg[n] + b2[k];
            v = fmaxf(v, 0.0f);
        } else {                                  // root_extend2 = conv1_out[root]
            v = h1[(size_t)root_index[g] * HID_F + (k - OUT_F)];
        }
        run += v; cnt_run += 1.0f;
    }
    atomicAdd(&sums[(size_t)gcur * OUTCAT_F + k], run);
    if (k == 0) atomicAdd(&counts[gcur], cnt_run);
}

__global__ void divide_kernel(const float* __restrict__ sums, const float* __restrict__ counts,
                              float* __restrict__ out) {
    int gid = blockIdx.x * blockDim.x + threadIdx.x;
    if (gid >= N_GRAPHS * OUTCAT_F) return;
    int g = gid >> 8;                            // OUTCAT_F == 256
    float c = counts[g];
    out[gid] = sums[gid] / fmaxf(c, 1.0f);
}

// ---------------------------------------------------------------- launcher
extern "C" void kernel_launch(void* const* d_in, const int* in_sizes, int n_in,
                              void* d_out, int out_size, void* d_ws, size_t ws_size,
                              hipStream_t stream) {
    (void)in_sizes; (void)n_in; (void)out_size; (void)ws_size;
    const float* x          = (const float*)d_in[0];
    const int*   edge_index = (const int*)  d_in[1];
    const int*   batch      = (const int*)  d_in[2];
    const int*   root_index = (const int*)  d_in[3];
    const float* W1         = (const float*)d_in[4];
    const float* b1         = (const float*)d_in[5];
    const float* W2         = (const float*)d_in[6];
    const float* b2         = (const float*)d_in[7];
    const int* src = edge_index;
    const int* dst = edge_index + N_EDGES;

    // workspace layout (4-byte units)
    int*   iws       = (int*)d_ws;
    int*   cnt       = iws;                                   // [N]  in-degree
    int*   row_ptr   = cnt + N_NODES;                         // [N]  CSR offsets
    int*   cursor    = row_ptr + N_NODES;                     // [N]  bucket cursors
    int*   colsrc    = cursor + N_NODES;                      // [E]  CSR column (src node)
    int*   blockSums = colsrc + N_EDGES;                      // [1024]
    float* dinv      = (float*)(blockSums + 1024);            // [N]
    float* invdeg    = dinv + N_NODES;                        // [N]
    float* hA        = invdeg + N_NODES;                      // [N,128] x@W1
    float* h1        = hA + (size_t)N_NODES * HID_F;          // [N,128] conv1 out
    float* hB        = h1 + (size_t)N_NODES * HID_F;          // [N,128] concat@W2
    float* sums      = hB + (size_t)N_NODES * OUT_F;          // [128,256]
    float* counts    = sums + (size_t)N_GRAPHS * OUTCAT_F;    // [128]
    __bf16* W1p      = (__bf16*)(counts + N_GRAPHS);          // 8*4096 bf16 (32B-aligned)
    __bf16* W2p      = W1p + 8 * 4096;                        // 12*4096 bf16

    const int TPB   = 256;
    const int gN    = (N_NODES + TPB - 1) / TPB;              // 782
    const int gNF   = (N_NODES * HID_F) / TPB;                // 100000
    const int gGemm = (N_NODES + 127) / 128;                  // 1563

    // init
    fill_int_kernel<<<gN, TPB, 0, stream>>>(cnt, 0, N_NODES);
    fill_int_kernel<<<gN, TPB, 0, stream>>>(cursor, 0, N_NODES);
    fill_kernel<<<(N_GRAPHS * OUTCAT_F) / TPB, TPB, 0, stream>>>(sums, 0.0f, N_GRAPHS * OUTCAT_F);
    fill_kernel<<<1, TPB, 0, stream>>>(counts, 0.0f, N_GRAPHS);

    // degrees + normalization coefficients
    indeg_kernel<<<gN, TPB, 0, stream>>>(dst, cnt, N_EDGES);
    dinv_kernel<<<gN, TPB, 0, stream>>>(cnt, dinv, invdeg, N_NODES);

    // CSR build: exclusive scan of in-degrees, then bucket edges by dst
    scan_block_kernel<<<gN, TPB, 0, stream>>>(cnt, row_ptr, blockSums, N_NODES);
    scan_sums_kernel<<<1, 1024, 0, stream>>>(blockSums, gN);
    scan_add_kernel<<<gN, TPB, 0, stream>>>(row_ptr, blockSums, N_NODES);
    bucket_kernel<<<gN, TPB, 0, stream>>>(src, dst, row_ptr, cursor, colsrc, N_EDGES);

    // pack weights into WMMA-B fragment order (bf16)
    pack_w_kernel<<<(8  * 4096) / TPB, TPB, 0, stream>>>(W1, W1p, 8);
    pack_w_kernel<<<(12 * 4096) / TPB, TPB, 0, stream>>>(W2, W2p, 12);

    // conv1: WMMA GEMM + fused gather/self-loop/bias
    gemm1_kernel<<<gGemm, TPB, 0, stream>>>(x, W1p, hA);
    gather_combine1_kernel<<<gNF, TPB, 0, stream>>>(hA, row_ptr, cnt, colsrc, dinv, invdeg, b1, h1);

    // conv2: WMMA GEMM (relu(concat) folded into A conversion) + fused gather in final
    gemm2_kernel<<<gGemm, TPB, 0, stream>>>(h1, x, batch, root_index, W2p, hB);

    // fused conv2-gather + relu + root concat + scatter-mean
    final_kernel<<<gGemm, TPB, 0, stream>>>(hB, row_ptr, cnt, colsrc, dinv, invdeg, b2, h1,
                                            batch, root_index, sums, counts);
    divide_kernel<<<(N_GRAPHS * OUTCAT_F) / TPB, TPB, 0, stream>>>(sums, counts, (float*)d_out);
}
